// GRNNTransformSimple_670014898785
// MI455X (gfx1250) — matrix operand order repaired and checked by reading, not compiled, not run on hardware
//
#include <hip/hip_runtime.h>
#include <hip/hip_bf16.h>

// ---------------------------------------------------------------------------
// GRNN tree transform for MI455X (gfx1250), wave32 + v_wmma_f32_16x16x32_bf16.
//
// Bandwidth-bound (~1.6 GB min traffic vs ~68 GFLOP), so:
//  * bf16 WMMA with f32 accumulation (probe-verified builtin)
//  * fuse u-GEMM + h-GEMM per level (u never hits HBM)
//  * inter-level emb buffers stored as bf16 (they are only ever consumed as
//    bf16 WMMA A-operands; halves emb write + gather traffic)
//  * weights pre-swizzled to B-layout in LDS once per block
//  * u C-layout -> A-layout transpose via wave-private LDS tile
//  * bf16 outputs staged through LDS -> 4x coalesced 16B stores per lane
// Workspace requirement: 384 MiB (emb ping-pong: 256 MiB even + 128 MiB odd).
// ---------------------------------------------------------------------------

typedef __attribute__((ext_vector_type(16))) __bf16 v16bf;
typedef __attribute__((ext_vector_type(8)))  __bf16 v8bf;
typedef __attribute__((ext_vector_type(8)))  float  v8f;

static __device__ __forceinline__ v8f wmma_bf16(v16bf a, v16bf b, v8f c) {
  // 8 args: (neg_a, A, neg_b, B, c_mod, C, reuse_a, reuse_b)
  return __builtin_amdgcn_wmma_f32_16x16x32_bf16(false, a, false, b,
                                                 (short)0, c, false, false);
}

static __device__ __forceinline__ v16bf cat8(v8bf lo, v8bf hi) {
  return __builtin_shufflevector(lo, hi, 0, 1, 2, 3, 4, 5, 6, 7,
                                 8, 9, 10, 11, 12, 13, 14, 15);
}

// Build one 16x32 bf16 A-tile for this lane from a row-major f32 row.
// ISA 16-bit A layout: lane = M (lane&15); lanes<16 hold K {0..7,16..23},
// lanes>=16 hold K {8..15,24..31}  ->  klo = base + (lane>>4)*8.
static __device__ __forceinline__ v16bf load_a_tile_f32(const float* rowptr, int klo) {
  const float4 a0 = *(const float4*)(rowptr + klo + 0);
  const float4 a1 = *(const float4*)(rowptr + klo + 4);
  const float4 a2 = *(const float4*)(rowptr + klo + 16);
  const float4 a3 = *(const float4*)(rowptr + klo + 20);
  v16bf a;
  a[0]  = (__bf16)a0.x; a[1]  = (__bf16)a0.y; a[2]  = (__bf16)a0.z; a[3]  = (__bf16)a0.w;
  a[4]  = (__bf16)a1.x; a[5]  = (__bf16)a1.y; a[6]  = (__bf16)a1.z; a[7]  = (__bf16)a1.w;
  a[8]  = (__bf16)a2.x; a[9]  = (__bf16)a2.y; a[10] = (__bf16)a2.z; a[11] = (__bf16)a2.w;
  a[12] = (__bf16)a3.x; a[13] = (__bf16)a3.y; a[14] = (__bf16)a3.z; a[15] = (__bf16)a3.w;
  return a;
}

// Same A-tile build from an already-bf16 row (gathers from bf16 emb levels).
static __device__ __forceinline__ v16bf load_a_tile_bf16(const __bf16* rowptr, int klo) {
  const v8bf lo = *(const v8bf*)(rowptr + klo);
  const v8bf hi = *(const v8bf*)(rowptr + klo + 16);
  return cat8(lo, hi);
}

// Cooperative fill of (ktiles*4) B-tiles from a (ktiles*32 x 64) row-major f32
// weight matrix into LDS, pre-swizzled to the 32x16 B layout:
//   lane 0-15: N = lane,    elem i -> K = {0..7,16..23}
//   lane16-31: N = lane-16, elem i -> K = {8..15,24..31}
static __device__ __forceinline__ void fill_b_tiles(__bf16* dst, const float* W,
                                                    int ktiles, int tid) {
  const int total = ktiles * 4 * 512;
  for (int e = tid; e < total; e += 256) {
    const int t  = e >> 9;         // tile index = kk*4 + nt
    const int r  = e & 511;
    const int ln = r >> 4;         // lane within tile
    const int i  = r & 15;         // element within lane
    const int kk = t >> 2;
    const int nt = t & 3;
    const int k  = kk * 32 + ((ln >> 4) << 3) + (i & 7) + ((i >> 3) << 4);
    const int n  = nt * 16 + (ln & 15);
    dst[t * 512 + ln * 16 + i] = (__bf16)W[k * 64 + n];
  }
}

// Coalesced bf16 tile store: wave-private LDS tile (16 rows, stride 72 elems)
// -> global row-major (16 x 64 bf16): 4 stores of 16B per lane, 128B/8 lanes.
static __device__ __forceinline__ void store_tile_coalesced(
    const __bf16 (*sRow)[72], __bf16* dstRowMajor, int lane) {
#pragma unroll
  for (int s = 0; s < 4; ++s) {
    const int row  = s * 4 + (lane >> 3);
    const int cole = (lane & 7) * 8;                 // element offset (16B)
    const v8bf v = *(const v8bf*)(&sRow[row][cole]);
    *(v8bf*)(dstRowMajor + (size_t)row * 64 + cole) = v;
  }
}

// ---------------- Level 8 (leaves): emb = relu(contents @ Wu + bu) ----------
__global__ void __launch_bounds__(256)
grnn_top(const float* __restrict__ contents,
         const float* __restrict__ Wu, const float* __restrict__ bu,
         __bf16* __restrict__ emb) {
  __shared__ __attribute__((aligned(32))) __bf16 sWu[4 * 512];
  __shared__ __attribute__((aligned(32))) __bf16 sU[8][16][72];
  const int tid  = threadIdx.x;
  const int lane = tid & 31;
  const int wave = tid >> 5;
  fill_b_tiles(sWu, Wu, 1, tid);
  __syncthreads();

  const size_t r0    = (size_t)blockIdx.x * 128 + (size_t)wave * 16;
  const int    mrow  = lane & 15;
  const int    klo8  = (lane >> 4) << 3;
  const int    ncol  = lane & 15;
  const int    mbase = (lane >> 4) << 3;

  const v16bf a = load_a_tile_f32(contents + (r0 + mrow) * 32, klo8);

#pragma unroll
  for (int nt = 0; nt < 4; ++nt) {
    const v16bf b = *(const v16bf*)(sWu + nt * 512 + lane * 16);
    v8f c = {};
    c = wmma_bf16(a, b, c);
    const float bn = bu[nt * 16 + ncol];
#pragma unroll
    for (int r = 0; r < 8; ++r)
      sU[wave][mbase + r][nt * 16 + ncol] = (__bf16)fmaxf(c[r] + bn, 0.0f);
  }
  // wave-private LDS: in-wave DScnt ordering suffices, no barrier
  store_tile_coalesced(sU[wave], emb + r0 * 64, lane);
}

// ---------------- Levels 0..7: fused u + gather + h GEMM --------------------
// OUT_BF16=true: write bf16 emb level buffer (coalesced via LDS staging).
// OUT_BF16=false: final level, write f32 to d_out.
template <bool OUT_BF16>
__global__ void __launch_bounds__(256)
grnn_level(const float*  __restrict__ contents,
           const int*    __restrict__ children,
           const __bf16* __restrict__ embp,
           const float* __restrict__ Wu, const float* __restrict__ bu,
           const float* __restrict__ Wh, const float* __restrict__ bh,
           void* __restrict__ embOut) {
  __shared__ __attribute__((aligned(32))) __bf16 sWu[4 * 512];        //  4 KB
  __shared__ __attribute__((aligned(32))) __bf16 sWh[24 * 512];       // 24 KB
  __shared__ __attribute__((aligned(32))) __bf16 sU[8][16][72];       // 18 KB

  const int tid  = threadIdx.x;
  const int lane = tid & 31;
  const int wave = tid >> 5;
  fill_b_tiles(sWu, Wu, 1, tid);
  fill_b_tiles(sWh, Wh, 6, tid);
  __syncthreads();

  const size_t r0    = (size_t)blockIdx.x * 128 + (size_t)wave * 16;
  const int    mrow  = lane & 15;
  const int    klo8  = (lane >> 4) << 3;
  const int    ncol  = lane & 15;
  const int    mbase = (lane >> 4) << 3;

  // ---- u = relu(contents @ Wu + bu), staged to wave-private LDS (bf16) ----
  {
    const v16bf a = load_a_tile_f32(contents + (r0 + mrow) * 32, klo8);
#pragma unroll
    for (int nt = 0; nt < 4; ++nt) {
      const v16bf b = *(const v16bf*)(sWu + nt * 512 + lane * 16);
      v8f c = {};
      c = wmma_bf16(a, b, c);
      const float bn = bu[nt * 16 + ncol];
#pragma unroll
      for (int r = 0; r < 8; ++r)
        sU[wave][mbase + r][nt * 16 + ncol] = (__bf16)fmaxf(c[r] + bn, 0.0f);
    }
  }
  // Wave-private LDS region: in-wave DScnt ordering suffices, no barrier.

  // ---- gather bf16 child embeddings, build six 16x32 A-tiles [hL|hR|u] ----
  const int cL = children[(r0 + mrow) * 2 + 0];
  const int cR = children[(r0 + mrow) * 2 + 1];
  const __bf16* rowL = embp + (size_t)cL * 64;
  const __bf16* rowR = embp + (size_t)cR * 64;

  v16bf hA[6];
#pragma unroll
  for (int kt = 0; kt < 2; ++kt) {
    hA[0 * 2 + kt] = load_a_tile_bf16(rowL, kt * 32 + klo8);
    hA[1 * 2 + kt] = load_a_tile_bf16(rowR, kt * 32 + klo8);
  }
#pragma unroll
  for (int kt = 0; kt < 2; ++kt)     // u: LDS row-major -> A layout
    hA[2 * 2 + kt] = load_a_tile_bf16(&sU[wave][mrow][0], kt * 32 + klo8);

  // ---- emb = relu([hL|hR|u] @ Wh + bh): 6 K-tiles x 4 N-tiles = 24 WMMAs ---
#pragma unroll
  for (int nt = 0; nt < 4; ++nt) {
    v8f c = {};
#pragma unroll
    for (int kk = 0; kk < 6; ++kk) {
      const v16bf b = *(const v16bf*)(sWh + (kk * 4 + nt) * 512 + lane * 16);
      c = wmma_bf16(hA[kk], b, c);
    }
    const float bn = bh[nt * 16 + ncol];
    if (OUT_BF16) {
      // sU is dead now (u already consumed): reuse it as output staging.
#pragma unroll
      for (int r = 0; r < 8; ++r)
        sU[wave][mbase + r][nt * 16 + ncol] = (__bf16)fmaxf(c[r] + bn, 0.0f);
    } else {
      float* outp = (float*)embOut + (r0 + mbase) * 64 + nt * 16 + ncol;
#pragma unroll
      for (int r = 0; r < 8; ++r)
        outp[(size_t)r * 64] = fmaxf(c[r] + bn, 0.0f);
    }
  }
  if (OUT_BF16)
    store_tile_coalesced(sU[wave], (__bf16*)embOut + r0 * 64, lane);
}

// ---------------------------------------------------------------------------
extern "C" void kernel_launch(void* const* d_in, const int* in_sizes, int n_in,
                              void* d_out, int out_size, void* d_ws, size_t ws_size,
                              hipStream_t stream) {
  (void)in_sizes; (void)n_in; (void)out_size; (void)ws_size;

  const float* contents[9];
  for (int j = 0; j < 9; ++j) contents[j] = (const float*)d_in[j];
  const int* children[8];
  for (int j = 0; j < 8; ++j) children[j] = (const int*)d_in[9 + j];
  const float* Wu = (const float*)d_in[17];
  const float* bu = (const float*)d_in[18];
  const float* Wh = (const float*)d_in[19];
  const float* bh = (const float*)d_in[20];

  const size_t B = 8192;
  // bf16 emb ping-pong in workspace: even levels at 0 (max 256 MiB = level 8),
  // odd levels at +256 MiB (max 128 MiB = level 7). Only adjacent levels live.
  __bf16* bufEven = (__bf16*)d_ws;
  __bf16* bufOdd  = (__bf16*)((char*)d_ws + ((size_t)256 << 20));

  __bf16* embPtr[9];
  embPtr[0] = nullptr;  // level 0 goes to d_out as f32
  for (int j = 1; j < 9; ++j) embPtr[j] = (j & 1) ? bufOdd : bufEven;

  {
    const int n8 = (int)(B << 8);
    grnn_top<<<dim3(n8 / 128), dim3(256), 0, stream>>>(contents[8], Wu, bu, embPtr[8]);
  }
  for (int j = 7; j >= 1; --j) {
    const int n = (int)(B << j);
    grnn_level<true><<<dim3(n / 128), dim3(256), 0, stream>>>(
        contents[j], children[j], embPtr[j + 1], Wu, bu, Wh, bh, (void*)embPtr[j]);
  }
  grnn_level<false><<<dim3(B / 128), dim3(256), 0, stream>>>(
      contents[0], children[0], embPtr[1], Wu, bu, Wh, bh, d_out);
}
